// VanillaLSTMCell_20177756357192
// MI455X (gfx1250) — compile-verified, matrix operand order checked by reading, not run on hardware
//
#include <hip/hip_runtime.h>

typedef __attribute__((ext_vector_type(16))) _Float16 v16h;
typedef __attribute__((ext_vector_type(8)))  float    v8f;
typedef __attribute__((ext_vector_type(4)))  float    v4f;

#define UNITS      256
#define INPUT_DIM  256
#define KTOT       (INPUT_DIM + UNITS)   // 512
#define NTOT       (4 * UNITS)           // 1024
#define KSTEPS     (KTOT / 32)           // 16

// ---------------------------------------------------------------------------
// Kernel 0: convert W [512,1024] f32 (K-major) -> Wt [1024,512] f16 (N-major).
// ---------------------------------------------------------------------------
__global__ __launch_bounds__(256) void lstm_convert_w(const float* __restrict__ W,
                                                      _Float16* __restrict__ Wt) {
    int idx = blockIdx.x * 256 + threadIdx.x;      // 0 .. 512*1024-1
    int k = idx >> 10;                             // K row of W
    int n = idx & 1023;                            // N col of W
    Wt[(size_t)n * KTOT + k] = (_Float16)W[idx];
}

// ---------------------------------------------------------------------------
// Fused LSTM cell. Block: 256 threads = 8 waves; wave w owns rows
// [m0+16w, m0+16w+16) and the same 16 gate-columns for all four gates.
// Grid: (UNITS/16, B/128)  -- column tiles vary fastest so the A row tile
// stays hot in L2 across all 16 column blocks.
// W fragments for the whole block are async-staged into LDS once, scattered
// directly into WMMA per-lane fragment order.
// ---------------------------------------------------------------------------
__device__ __forceinline__ float sigmoid_f(float v) {
    return 1.0f / (1.0f + __expf(-v));
}

__device__ __forceinline__ v16h load_a_frag(const float* __restrict__ src,
                                            int rowA, int kb, int koff) {
    // A 16x32 f16 layout: lane<16 -> K runs {kb+0..7, kb+16..23};
    //                     lane>=16 -> K runs {kb+8..15, kb+24..31}
    const float* p = src + (size_t)rowA * INPUT_DIM + kb + koff;
    v4f r0 = ((const v4f*)p)[0];
    v4f r1 = ((const v4f*)p)[1];
    v4f r2 = ((const v4f*)(p + 16))[0];
    v4f r3 = ((const v4f*)(p + 16))[1];
    v16h a;
#pragma unroll
    for (int j = 0; j < 4; ++j) {
        a[j]      = (_Float16)r0[j];
        a[4 + j]  = (_Float16)r1[j];
        a[8 + j]  = (_Float16)r2[j];
        a[12 + j] = (_Float16)r3[j];
    }
    return a;
}

#define WMMA_F16(acc, a, b) \
    __builtin_amdgcn_wmma_f32_16x16x32_f16(false, (a), false, (b), (short)0, (acc), false, false)

__global__ __launch_bounds__(256) void lstm_fused(
        const float* __restrict__ x,
        const float* __restrict__ h_prev,
        const float* __restrict__ c_prev,
        const _Float16* __restrict__ Wt,       // [1024][512] f16, N-major
        const float* __restrict__ bias,
        const float* __restrict__ peep_i,
        const float* __restrict__ peep_f,
        const float* __restrict__ peep_o,
        float* __restrict__ out,               // [h | h | c], each B*UNITS f32
        int Bsz) {
    // B fragments in final WMMA order: [gate][kstep][lane][16 halves] = 64 KB
    __shared__ _Float16 smemB[4 * KSTEPS * 32 * 16];

    const int tid  = threadIdx.x;
    const int lane = tid & 31;
    const int wave = tid >> 5;
    const int n0   = blockIdx.x * 16;                // gate-column tile
    const int m0   = blockIdx.y * 128 + wave * 16;   // wave's row tile
    const int koff = (lane & 16) ? 8 : 0;
    const int rowA = m0 + (lane & 15);
    const int col  = n0 + (lane & 15);

    // ---- async-stage W fragments: global -> LDS, scattered into frag order ----
    {
        const uint32_t lds0 = (uint32_t)(uintptr_t)(void*)smemB;
#pragma unroll
        for (int s = 0; s < 16; ++s) {
            const int u     = s * 256 + tid;          // 16B unit id, 0..4095
            const int half  = u & 1;                  // which 8-half run
            const int bl    = (u >> 1) & 31;          // fragment lane
            const int kstep = (u >> 6) & 15;
            const int g     = u >> 10;                // gate
            const int gcol  = g * UNITS + n0 + (bl & 15);
            const int kk    = kstep * 32 + ((bl >> 4) << 3) + half * 16;
            const uint64_t ga = (uint64_t)(uintptr_t)(Wt + ((size_t)gcol * KTOT + kk));
            const uint32_t la = lds0 +
                (uint32_t)((((g * KSTEPS + kstep) * 32 + bl) * 16 + half * 8) * 2);
            asm volatile("global_load_async_to_lds_b128 %0, %1, off"
                         :: "v"(la), "v"(ga) : "memory");
        }
#if __has_builtin(__builtin_amdgcn_s_wait_asynccnt)
        __builtin_amdgcn_s_wait_asynccnt(0);
#else
        asm volatile("s_wait_asynccnt 0x0" ::: "memory");
#endif
        __syncthreads();
    }

#define LDB(G, KS) (*(const v16h*)&smemB[(((G) * KSTEPS + (KS)) * 32 + lane) * 16])

    v8f acc0 = {0.f, 0.f, 0.f, 0.f, 0.f, 0.f, 0.f, 0.f};   // i gate
    v8f acc1 = acc0;                                        // f gate
    v8f acc2 = acc0;                                        // c_hat
    v8f acc3 = acc0;                                        // o gate

    // ---- K loop, first half: A = x ----
#pragma unroll 2
    for (int ks = 0; ks < 8; ++ks) {
        v16h a = load_a_frag(x, rowA, ks * 32, koff);
        acc0 = WMMA_F16(acc0, a, LDB(0, ks));
        acc1 = WMMA_F16(acc1, a, LDB(1, ks));
        acc2 = WMMA_F16(acc2, a, LDB(2, ks));
        acc3 = WMMA_F16(acc3, a, LDB(3, ks));
    }
    // ---- K loop, second half: A = h_prev ----
#pragma unroll 2
    for (int ks = 8; ks < 16; ++ks) {
        v16h a = load_a_frag(h_prev, rowA, (ks - 8) * 32, koff);
        acc0 = WMMA_F16(acc0, a, LDB(0, ks));
        acc1 = WMMA_F16(acc1, a, LDB(1, ks));
        acc2 = WMMA_F16(acc2, a, LDB(2, ks));
        acc3 = WMMA_F16(acc3, a, LDB(3, ks));
    }
#undef LDB

    // ---- Epilogue: peepholes + nonlinearities, all in registers ----
    const float bi = bias[0 * UNITS + col];
    const float bf = bias[1 * UNITS + col];
    const float bc = bias[2 * UNITS + col];
    const float bo = bias[3 * UNITS + col];
    const float pi = peep_i[col];
    const float pf = peep_f[col];
    const float po = peep_o[col];

    const size_t N     = (size_t)Bsz * UNITS;
    const int    rbase = m0 + ((lane & 16) ? 8 : 0);   // C layout: VGPR v -> row v (+8)

#pragma unroll
    for (int v = 0; v < 8; ++v) {
        const int    row = rbase + v;
        const size_t off = (size_t)row * UNITS + col;
        const float  cp  = c_prev[off];

        const float ig = sigmoid_f(acc0[v] + bi + pi * cp);
        const float fg = sigmoid_f(acc1[v] + bf + pf * cp);
        const float og = sigmoid_f(acc3[v] + bo + po * cp);
        const float ch = tanhf(acc2[v] + bc);

        const float c = fg * cp + ig * ch;
        const float h = og * tanhf(c);

        out[off]         = h;
        out[N + off]     = h;
        out[2 * N + off] = c;
    }
}

// ---------------------------------------------------------------------------
extern "C" void kernel_launch(void* const* d_in, const int* in_sizes, int n_in,
                              void* d_out, int out_size, void* d_ws, size_t ws_size,
                              hipStream_t stream) {
    const float* x      = (const float*)d_in[0];
    const float* h_prev = (const float*)d_in[1];
    const float* c_prev = (const float*)d_in[2];
    const float* W      = (const float*)d_in[3];
    const float* bias   = (const float*)d_in[4];
    const float* pi     = (const float*)d_in[5];
    const float* pf     = (const float*)d_in[6];
    const float* po     = (const float*)d_in[7];

    const int Bsz = in_sizes[0] / INPUT_DIM;    // 131072

    _Float16* Wt = (_Float16*)d_ws;             // 1 MB of scratch

    // 1) transpose+convert W -> f16 [1024][512]
    lstm_convert_w<<<(KTOT * NTOT) / 256, 256, 0, stream>>>(W, Wt);

    // 2) fused WMMA GEMM + LSTM epilogue; column tiles vary fastest for L2 reuse
    dim3 grid(UNITS / 16, Bsz / 128);
    lstm_fused<<<grid, 256, 0, stream>>>(x, h_prev, c_prev, Wt, bias,
                                         pi, pf, po, (float*)d_out, Bsz);
}
